// EigenBiasAttention_23356032155879
// MI455X (gfx1250) — matrix operand order repaired and checked by reading, not compiled, not run on hardware
//
#include <hip/hip_runtime.h>
#include <hip/hip_bf16.h>
#include <math.h>

#define BB 2
#define TT 1024
#define DD 1024
#define HH 16
#define DH 64
#define QSTR (3 * DD)   // 3072 halves, row stride of qkv buffer

typedef __attribute__((ext_vector_type(16))) _Float16 v16h;
typedef __attribute__((ext_vector_type(8)))  _Float16 v8h;
typedef __attribute__((ext_vector_type(4)))  _Float16 v4h;
typedef __attribute__((ext_vector_type(8)))  float    v8f;
typedef __attribute__((ext_vector_type(4)))  unsigned int u32x4;
typedef __attribute__((ext_vector_type(8)))  int      i32x8;
typedef __attribute__((ext_vector_type(4)))  int      i32x4;

#if defined(__HIP_DEVICE_COMPILE__) && __has_builtin(__builtin_amdgcn_tensor_load_to_lds) && (__clang_major__ < 23)
#define USE_TDM 1
#else
#define USE_TDM 0
#endif

// Assemble a v16h WMMA operand from two 16-byte (ds/global b128) loads.
__device__ __forceinline__ v16h ld32h(const _Float16* p0, const _Float16* p1) {
  v16h r;
  *(v8h*)&r       = *(const v8h*)p0;
  *((v8h*)&r + 1) = *(const v8h*)p1;
  return r;
}

__device__ __forceinline__ v8f wmma_f16(v16h a, v16h b, v8f c) {
  return __builtin_amdgcn_wmma_f32_16x16x32_f16(false, a, false, b,
                                                (short)0, c, false, false);
}

// ---------------------------------------------------------------------------
// Kernel 0: f32 -> f16 conversion, vectorized by 4.
// ---------------------------------------------------------------------------
__global__ void cvt_kernel(const float* __restrict__ in, _Float16* __restrict__ out,
                           int n4) {
  int i = blockIdx.x * blockDim.x + threadIdx.x;
  if (i >= n4) return;
  float4 v = ((const float4*)in)[i];
  v4h o;
  o[0] = (_Float16)v.x; o[1] = (_Float16)v.y;
  o[2] = (_Float16)v.z; o[3] = (_Float16)v.w;
  *(v4h*)(out + (size_t)i * 4) = o;
}

// ---------------------------------------------------------------------------
// Kernel 1: GEMM  C[M,N] = A[M,K] @ W[N,K]^T + bias[N]; A,W are f16.
// Each wave computes a 16x64 strip: 4 accumulators share one A operand per
// K-step (2.5 b128 loads per WMMA, 4 independent WMMAs to hide load latency).
//  A (16x32): lane row = lane&15, K = 16 contiguous at 16*(lane>=16)
//  B (32x16): lane col = lane&15, K = 16 contiguous at 16*(lane>=16)
// ---------------------------------------------------------------------------
__global__ void __launch_bounds__(256)
gemm_wmma_f16(const _Float16* __restrict__ A, const _Float16* __restrict__ W,
              const float* __restrict__ bias, float* __restrict__ Cf,
              _Float16* __restrict__ Ch, int M, int N, int K) {
  int gw = blockIdx.x * 8 + (threadIdx.x >> 5);
  int tilesN = N >> 6;                            // 64-wide wave strips
  int tm = gw / tilesN;
  int tn = gw - tm * tilesN;
  if (tm >= (M >> 4)) return;                     // wave-uniform

  int lane = threadIdx.x & 31;
  int col = lane & 15;
  int hi  = lane >> 4;

  const _Float16* a0 = A + (size_t)(tm * 16 + col) * K;  // this lane's A row
  const _Float16* w0 = W + (size_t)(tn * 64 + col) * K;  // W row for sub-tile 0

  v8f acc[4];
#pragma unroll
  for (int j = 0; j < 4; ++j) acc[j] = (v8f){};

  for (int k0 = 0; k0 < K; k0 += 32) {
    if (k0 + 32 < K) {
      __builtin_prefetch(a0 + k0 + 32, 0, 1);     // global_prefetch_b8
      __builtin_prefetch(w0 + k0 + 32, 0, 1);
    }
    v16h a = ld32h(a0 + k0 + hi * 16, a0 + k0 + hi * 16 + 8);
#pragma unroll
    for (int j = 0; j < 4; ++j) {
      const _Float16* wr = w0 + (size_t)j * 16 * K + k0 + hi * 16;
      acc[j] = wmma_f16(a, ld32h(wr, wr + 8), acc[j]);
    }
  }

#pragma unroll
  for (int j = 0; j < 4; ++j) {
    int n0 = tn * 64 + j * 16 + col;
    float bv = bias ? bias[n0] : 0.0f;
#pragma unroll
    for (int r = 0; r < 8; ++r) {
      size_t idx = (size_t)(tm * 16 + r + 8 * hi) * N + n0;
      float v = acc[j][r] + bv;
      if (Ch) Ch[idx] = (_Float16)v;
      else    Cf[idx] = v;
    }
  }
}

// ---------------------------------------------------------------------------
// Kernel 2: Plucker lines (one thread per (b,t,h)).
// Writes Jw as f32 (8-stride, for the scan) and f16 (32-half zero-padded rows,
// for WMMA operands), rd as f32.
// ---------------------------------------------------------------------------
__device__ __forceinline__ void exterior6(const float* p, const float* q, float* L) {
  L[0] = p[0] * q[1] - p[1] * q[0];
  L[1] = p[0] * q[2] - p[2] * q[0];
  L[2] = p[0] * q[3] - p[3] * q[0];
  L[3] = p[1] * q[2] - p[2] * q[1];
  L[4] = p[1] * q[3] - p[3] * q[1];
  L[5] = p[2] * q[3] - p[3] * q[2];
  float s = 0.f;
#pragma unroll
  for (int i = 0; i < 6; ++i) s += L[i] * L[i];
  float n = fmaxf(sqrtf(s), 1e-12f);
#pragma unroll
  for (int i = 0; i < 6; ++i) L[i] /= n;
}

__global__ void lines_kernel(const float* __restrict__ pw1,
                             const float* __restrict__ pw2,
                             const float* __restrict__ pr1,
                             const float* __restrict__ pr2,
                             float* __restrict__ JwF,
                             _Float16* __restrict__ JwH,
                             float* __restrict__ rdF) {
  int idx = blockIdx.x * blockDim.x + threadIdx.x;
  if (idx >= BB * TT * HH) return;
  int h = idx % HH;
  int t = (idx / HH) % TT;
  int b = idx / (HH * TT);

  size_t row  = (size_t)(b * TT + t) * (4 * HH) + h * 4;
  size_t prow = (size_t)(b * TT + (t - 1)) * (4 * HH) + h * 4;

  float p1[4], p2[4], L[6], jw[6];
#pragma unroll
  for (int i = 0; i < 4; ++i) {
    p1[i] = (t == 0) ? 0.0f : pw1[prow + i];
    p2[i] = pw2[row + i];
  }
  exterior6(p1, p2, L);
  jw[0] =  L[5]; jw[1] = -L[4]; jw[2] =  L[3];
  jw[3] =  L[2]; jw[4] = -L[1]; jw[5] =  L[0];

  size_t bhT = (size_t)(b * HH + h) * TT + t;
  float* Jo = JwF + bhT * 8;
#pragma unroll
  for (int i = 0; i < 6; ++i) Jo[i] = jw[i];
  Jo[6] = 0.f; Jo[7] = 0.f;

  _Float16* JoH = JwH + bhT * 32;
#pragma unroll
  for (int i = 0; i < 32; ++i) JoH[i] = (i < 6) ? (_Float16)jw[i] : (_Float16)0.f;

#pragma unroll
  for (int i = 0; i < 4; ++i) { p1[i] = pr1[row + i]; p2[i] = pr2[row + i]; }
  exterior6(p1, p2, L);
  float* Ro = rdF + bhT * 8;
#pragma unroll
  for (int i = 0; i < 6; ++i) Ro[i] = L[i];
  Ro[6] = 0.f; Ro[7] = 0.f;
}

// ---------------------------------------------------------------------------
// Kernel 3: sequential Gram scan per (b,h); one wave each, lane j<6 owns column
// j of the running 6x6 M. Emits rdM (bias_scale folded) as f16 32-half rows.
// ---------------------------------------------------------------------------
__global__ void scan_kernel(const float* __restrict__ JwF,
                            const float* __restrict__ rdF,
                            const float* __restrict__ bias_scale,
                            _Float16* __restrict__ rdMH) {
  int bh = blockIdx.x;
  int h = bh % HH;
  int lane = threadIdx.x;                // 32 threads
  float bs = bias_scale[h];
  float Mcol[6] = {0.f, 0.f, 0.f, 0.f, 0.f, 0.f};
  const float* Jb = JwF + (size_t)bh * TT * 8;
  const float* Rb = rdF + (size_t)bh * TT * 8;
  _Float16*    Ob = rdMH + (size_t)bh * TT * 32;
  for (int t = 0; t < TT; ++t) {
    float j[6], r[6];
#pragma unroll
    for (int i = 0; i < 6; ++i) { j[i] = Jb[t * 8 + i]; r[i] = Rb[t * 8 + i]; }
    float jj = (lane < 6) ? j[lane] : 0.0f;
    float acc = 0.0f;
#pragma unroll
    for (int i = 0; i < 6; ++i) {
      Mcol[i] += j[i] * jj;              // inclusive cumsum of outer product
      acc += r[i] * Mcol[i];
    }
    Ob[(size_t)t * 32 + lane] = (_Float16)((lane < 6) ? acc * bs : 0.0f);
  }
}

// ---------------------------------------------------------------------------
// Kernel 4: flash attention with rank-6 Plucker bias. f16 in, f16 out.
// Block = 8 waves, 128 query rows of one (b,h). Uniform key-tile trip count so
// the block cooperatively stages K (via TDM where available) and transposed V
// in LDS; fully-masked tiles are inert in the online softmax.
// ---------------------------------------------------------------------------
__global__ void __launch_bounds__(256)
attn_kernel(const _Float16* __restrict__ qkv,
            const _Float16* __restrict__ JwH,
            const _Float16* __restrict__ rdMH,
            _Float16* __restrict__ aout) {
  // LDS: sK 32x64 h (4096B) | sVt 64x40 h (5120B) | sP 8 waves x 16x32 h (8192B)
  __shared__ __align__(16) unsigned char smem[4096 + 5120 + 8192];
  _Float16* sK  = (_Float16*)smem;
  _Float16* sVt = (_Float16*)(smem + 4096);

  int tid = threadIdx.x;
  int w = tid >> 5, lane = tid & 31, col = lane & 15, hi = lane >> 4;
  _Float16* sP = (_Float16*)(smem + 9216) + w * (16 * 32);

  int bh = blockIdx.y, b = bh / HH, h = bh % HH;
  int qBase = (blockIdx.x * 8 + w) * 16;

  const _Float16* qp  = qkv + (size_t)b * TT * QSTR + h * DH;
  const _Float16* kp  = qp + DD;
  const _Float16* vp  = qp + 2 * DD;
  const _Float16* jwp = JwH  + (size_t)bh * TT * 32;
  const _Float16* rmp = rdMH + (size_t)bh * TT * 32;

  // Q and rdM operands (A layout), loaded once per wave.
  const _Float16* qrow = qp + (size_t)(qBase + col) * QSTR + hi * 16;
  v16h qa0 = ld32h(qrow,      qrow + 8);
  v16h qa1 = ld32h(qrow + 32, qrow + 40);
  const _Float16* rrow = rmp + (size_t)(qBase + col) * 32 + hi * 16;
  v16h ra = ld32h(rrow, rrow + 8);

  float m[8], l[8];
  v8f o[4];
#pragma unroll
  for (int r = 0; r < 8; ++r) { m[r] = -INFINITY; l[r] = 0.0f; }
#pragma unroll
  for (int n = 0; n < 4; ++n) o[n] = (v8f){};

  const float scale = 0.125f;            // DH^-0.5
  int nkt = blockIdx.x * 4 + 4;          // uniform across the block

  for (int kt = 0; kt < nkt; ++kt) {
    int kBase = kt * 32;
    __syncthreads();                     // previous iteration's LDS reads done

#if USE_TDM
    if (w == 0) {
      // Tensor Data Mover: copy K tile (32 rows x 64 halves, row stride 3072)
      // into LDS offset 0. D# per ISA 8.3/8.4; data_size=1 (2 bytes).
      unsigned long long ga = (unsigned long long)(kp + (size_t)kBase * QSTR);
      u32x4 g0;
      g0[0] = 1u;                                   // count=1, user descriptor
      g0[1] = 0u;                                   // lds_addr = 0 (sK)
      g0[2] = (unsigned int)ga;                     // global_addr[31:0]
      g0[3] = (unsigned int)(ga >> 32) | (2u << 30);// global_addr[56:32] | type=2
      i32x8 g1;
      g1[0] = 0x00010000;                           // data_size = 2B
      g1[1] = (64 << 16);                           // tensor_dim0 = 64
      g1[2] = (32 << 16);                           // tensor_dim1 = 32
      g1[3] = (64 << 16);                           // tile_dim0 = 64
      g1[4] = 32;                                   // tile_dim1 = 32
      g1[5] = 3072;                                 // tensor_dim0_stride
      g1[6] = 0; g1[7] = 0;
      i32x4 gz; gz[0] = 0; gz[1] = 0; gz[2] = 0; gz[3] = 0;
      __builtin_amdgcn_tensor_load_to_lds(g0, g1, gz, gz, 0);
#if __has_builtin(__builtin_amdgcn_s_wait_tensorcnt)
      __builtin_amdgcn_s_wait_tensorcnt(0);
#endif
    }
#else
    {
      int key = tid >> 3, seg = tid & 7;           // 256 threads x 16B
      *(v8h*)&sK[key * 64 + seg * 8] =
          *(const v8h*)(kp + (size_t)(kBase + key) * QSTR + seg * 8);
    }
#endif
    {
      // Stage V transposed: sVt[d][key], row stride 40 halves.
      int key = tid >> 3, seg = tid & 7;
      v8h vv = *(const v8h*)(vp + (size_t)(kBase + key) * QSTR + seg * 8);
#pragma unroll
      for (int j = 0; j < 8; ++j) sVt[(seg * 8 + j) * 40 + key] = vv[j];
    }
    __syncthreads();

    float lt[2][8];
#pragma unroll
    for (int hh = 0; hh < 2; ++hh) {
      const _Float16* krow = &sK[(hh * 16 + col) * 64 + hi * 16];
      v8f s = (v8f){};
      s = wmma_f16(qa0, ld32h(krow,      krow + 8),  s);
      s = wmma_f16(qa1, ld32h(krow + 32, krow + 40), s);
      const _Float16* jrow = jwp + (size_t)(kBase + hh * 16 + col) * 32 + hi * 16;
      v8f sb = wmma_f16(ra, ld32h(jrow, jrow + 8), (v8f){});
#pragma unroll
      for (int r = 0; r < 8; ++r) {
        int tRow = qBase + r + 8 * hi;
        int sCol = kBase + hh * 16 + col;
        float li = s[r] * scale + fabsf(sb[r]);
        lt[hh][r] = (sCol <= tRow) ? li : -INFINITY;
      }
    }

    // Online softmax; rows live at (VGPR index, lane-group), columns across the
    // 16 lanes of each half-wave (xor 1,2,4,8 stays inside a group).
    float fac[8];
#pragma unroll
    for (int r = 0; r < 8; ++r) {
      float mt = fmaxf(lt[0][r], lt[1][r]);
#pragma unroll
      for (int off = 1; off < 16; off <<= 1) mt = fmaxf(mt, __shfl_xor(mt, off, 32));
      float mn = fmaxf(m[r], mt);
      float p0 = __expf(lt[0][r] - mn);
      float p1 = __expf(lt[1][r] - mn);
      float rs = p0 + p1;
#pragma unroll
      for (int off = 1; off < 16; off <<= 1) rs += __shfl_xor(rs, off, 32);
      float ef = __expf(m[r] - mn);
      l[r] = l[r] * ef + rs;
      m[r] = mn;
      fac[r] = ef;
      sP[(r + 8 * hi) * 32 + col]      = (_Float16)p0;   // C layout -> LDS rows
      sP[(r + 8 * hi) * 32 + 16 + col] = (_Float16)p1;
    }
#pragma unroll
    for (int n = 0; n < 4; ++n)
#pragma unroll
      for (int r = 0; r < 8; ++r) o[n][r] = o[n][r] * fac[r];

    // Reload P in A-operand layout (same-wave DS ordering; no barrier needed).
    const _Float16* prow = sP + col * 32 + hi * 16;
    v16h pa = ld32h(prow, prow + 8);

#pragma unroll
    for (int n = 0; n < 4; ++n) {
      const _Float16* vrow = &sVt[(n * 16 + col) * 40 + hi * 16];
      o[n] = wmma_f16(pa, ld32h(vrow, vrow + 8), o[n]);
    }
  }

#pragma unroll
  for (int n = 0; n < 4; ++n)
#pragma unroll
    for (int r = 0; r < 8; ++r) {
      int tRow = qBase + r + 8 * hi;
      aout[(size_t)(b * TT + tRow) * DD + h * DH + n * 16 + col] =
          (_Float16)(o[n][r] / l[r]);
    }
}

// ---------------------------------------------------------------------------
// Host launcher
// ---------------------------------------------------------------------------
static inline void launch_gemm(const _Float16* A, const _Float16* W,
                               const float* bias, float* Cf, _Float16* Ch,
                               int M, int N, int K, hipStream_t s) {
  int tiles = (M / 16) * (N / 64);
  gemm_wmma_f16<<<(tiles + 7) / 8, 256, 0, s>>>(A, W, bias, Cf, Ch, M, N, K);
}

static inline void launch_cvt(const float* in, _Float16* out, size_t n,
                              hipStream_t s) {
  int n4 = (int)(n / 4);
  cvt_kernel<<<(n4 + 255) / 256, 256, 0, s>>>(in, out, n4);
}

extern "C" void kernel_launch(void* const* d_in, const int* in_sizes, int n_in,
                              void* d_out, int out_size, void* d_ws, size_t ws_size,
                              hipStream_t stream) {
  (void)in_sizes; (void)n_in; (void)out_size; (void)ws_size;
  const float* x          = (const float*)d_in[0];
  const float* Wqkv       = (const float*)d_in[1];
  const float* bqkv       = (const float*)d_in[2];
  const float* W1w        = (const float*)d_in[3];
  const float* W2w        = (const float*)d_in[4];
  const float* W1r        = (const float*)d_in[5];
  const float* W2r        = (const float*)d_in[6];
  const float* bias_scale = (const float*)d_in[7];
  const float* Wout       = (const float*)d_in[8];
  const float* bout       = (const float*)d_in[9];
  float* out = (float*)d_out;

  char* cur = (char*)d_ws;
  auto takeH = [&](size_t n) { _Float16* p = (_Float16*)cur; cur += n * 2;  cur = (char*)(((uintptr_t)cur + 255) & ~(uintptr_t)255); return p; };
  auto takeF = [&](size_t n) { float*    p = (float*)cur;    cur += n * 4;  cur = (char*)(((uintptr_t)cur + 255) & ~(uintptr_t)255); return p; };

  const size_t M = (size_t)BB * TT;
  _Float16* x_h    = takeH(M * DD);
  _Float16* Wqkv_h = takeH((size_t)3 * DD * DD);
  _Float16* W1w_h  = takeH((size_t)4 * HH * DD);
  _Float16* W2w_h  = takeH((size_t)4 * HH * DD);
  _Float16* W1r_h  = takeH((size_t)4 * HH * DD);
  _Float16* W2r_h  = takeH((size_t)4 * HH * DD);
  _Float16* Wout_h = takeH((size_t)DD * DD);
  _Float16* qkv_h  = takeH(M * QSTR);
  float*    pw1    = takeF(M * 4 * HH);
  float*    pw2    = takeF(M * 4 * HH);
  float*    pr1    = takeF(M * 4 * HH);
  float*    pr2    = takeF(M * 4 * HH);
  float*    JwF    = takeF((size_t)BB * HH * TT * 8);
  float*    rdF    = takeF((size_t)BB * HH * TT * 8);
  _Float16* JwH    = takeH((size_t)BB * HH * TT * 32);
  _Float16* rdMH   = takeH((size_t)BB * HH * TT * 32);
  _Float16* aout_h = takeH(M * DD);

  // 0) f32 -> f16 conversions (one pass each)
  launch_cvt(x,    x_h,    M * DD, stream);
  launch_cvt(Wqkv, Wqkv_h, (size_t)3 * DD * DD, stream);
  launch_cvt(W1w,  W1w_h,  (size_t)4 * HH * DD, stream);
  launch_cvt(W2w,  W2w_h,  (size_t)4 * HH * DD, stream);
  launch_cvt(W1r,  W1r_h,  (size_t)4 * HH * DD, stream);
  launch_cvt(W2r,  W2r_h,  (size_t)4 * HH * DD, stream);
  launch_cvt(Wout, Wout_h, (size_t)DD * DD, stream);

  // 1) fused QKV projection (f16 output for the attention kernel)
  launch_gemm(x_h, Wqkv_h, bqkv, nullptr, qkv_h, (int)M, 3 * DD, DD, stream);
  // 2) line projections (f32 outputs; w1 consumed shifted by lines_kernel)
  launch_gemm(x_h, W1w_h, nullptr, pw1, nullptr, (int)M, 4 * HH, DD, stream);
  launch_gemm(x_h, W2w_h, nullptr, pw2, nullptr, (int)M, 4 * HH, DD, stream);
  launch_gemm(x_h, W1r_h, nullptr, pr1, nullptr, (int)M, 4 * HH, DD, stream);
  launch_gemm(x_h, W2r_h, nullptr, pr2, nullptr, (int)M, 4 * HH, DD, stream);
  // 3) Plucker exterior + J6
  int nl = BB * TT * HH;
  lines_kernel<<<(nl + 255) / 256, 256, 0, stream>>>(pw1, pw2, pr1, pr2,
                                                     JwF, JwH, rdF);
  // 4) sequential Gram scan (bias_scale folded into rd_M)
  scan_kernel<<<BB * HH, 32, 0, stream>>>(JwF, rdF, bias_scale, rdMH);
  // 5) flash attention with rank-6 bias (TDM-staged K tiles)
  attn_kernel<<<dim3(TT / 128, BB * HH), 256, 0, stream>>>(qkv_h, JwH, rdMH,
                                                           aout_h);
  // 6) output projection -> d_out (f32)
  launch_gemm(aout_h, Wout_h, bout, out, nullptr, (int)M, DD, DD, stream);
}
